// fA_P4MResBlock2D_55327768707259
// MI455X (gfx1250) — matrix operand order repaired and checked by reading, not compiled, not run on hardware
//
#include <hip/hip_runtime.h>

#define HW 4096
#define Bb 8
#define CIN8 512
#define NCHUNK 144   // 9 taps * 16 cin-blocks of 32
#define NPAIR  72    // K=64 double-chunks per (b,g) tile

typedef __attribute__((ext_vector_type(16))) __bf16 v16bf;
typedef __attribute__((ext_vector_type(8)))  float  v8f;

__device__ __forceinline__ unsigned short f2bf(float f) {
  unsigned u = __builtin_bit_cast(unsigned, f);
  return (unsigned short)((u + 0x7fffu + ((u >> 16) & 1u)) >> 16);
}
__device__ __forceinline__ float bf2f(unsigned short h) {
  unsigned u = ((unsigned)h) << 16;
  return __builtin_bit_cast(float, u);
}

// p4m: element i = (mirror = i>>2, rot = i&3). PERM[g][h] = idx(inv(g)*h)
__device__ __forceinline__ int perm_idx(int g, int h) {
  int ag = g >> 2, bg = g & 3, ah = h >> 2, bh = h & 3;
  int bi = (ag == 0) ? ((4 - bg) & 3) : bg;     // inv(g) = (ag, bi)
  int a = (ag + ah) & 1;
  int t = (ah == 0) ? bi : ((4 - bi) & 3);
  int b = (t + bh) & 3;
  return a * 4 + b;
}

// map transformed-kernel output coords (kh,kw) -> source coords (flip then rot90^b)
__device__ __forceinline__ void inv_spatial(int g, int kh, int kw, int N, int* sh, int* sw) {
  int a = g >> 2, b = g & 3;
  int h = kh, w = kw;
  for (int i = 0; i < b; i++) { int nh = w, nw = (N - 1) - h; h = nh; w = nw; }
  if (a) w = (N - 1) - w;
  *sh = h; *sw = w;
}

// ---------------- Stage 1: BN + ReLU -> bf16 channels-last, per-(b,c,g) mean/max
__global__ void bn_relu_stats_k(const float* __restrict__ x,
                                const float* __restrict__ gamma, const float* __restrict__ beta,
                                const float* __restrict__ mean,  const float* __restrict__ var,
                                unsigned short* __restrict__ yb,
                                float* __restrict__ avg, float* __restrict__ mx) {
  int blk = blockIdx.x;                 // (b*64+c)*8+g
  int g = blk & 7; int bc = blk >> 3; int c = bc & 63; int b = bc >> 6;
  float inv = gamma[c] * rsqrtf(var[c] + 2e-5f);
  float shf = beta[c] - mean[c] * inv;
  const float* xp = x + (size_t)blk * HW;
  unsigned short* yp = yb + (size_t)b * HW * CIN8 + (c * 8 + g);
  float s = 0.f, m = -1e30f;
  for (int p = threadIdx.x; p < HW; p += 256) {
    float v = xp[p] * inv + shf;
    v = v > 0.f ? v : 0.f;
    s += v; m = fmaxf(m, v);
    yp[(size_t)p * CIN8] = f2bf(v);
  }
  __shared__ float rs[256], rm[256];
  rs[threadIdx.x] = s; rm[threadIdx.x] = m; __syncthreads();
  for (int o = 128; o > 0; o >>= 1) {
    if (threadIdx.x < o) {
      rs[threadIdx.x] += rs[threadIdx.x + o];
      rm[threadIdx.x] = fmaxf(rm[threadIdx.x], rm[threadIdx.x + o]);
    }
    __syncthreads();
  }
  if (threadIdx.x == 0) { avg[blk] = rs[0] * (1.f / HW); mx[blk] = rm[0]; }
}

// ---------------- Stage 2: per-pixel mean/max over C -> f[b][2][g][pix]
__global__ void ch_meanmax_k(const unsigned short* __restrict__ yb, float* __restrict__ f) {
  int idx = blockIdx.x * 256 + threadIdx.x;   // over B*8*HW
  int p = idx & (HW - 1); int bg = idx >> 12; int g = bg & 7; int b = bg >> 3;
  const unsigned short* yp = yb + (size_t)(b * HW + p) * CIN8 + g;
  float s = 0.f, m = -1e30f;
  for (int c = 0; c < 64; c++) { float v = bf2f(yp[c * 8]); s += v; m = fmaxf(m, v); }
  f[((size_t)(b * 2 + 0) * 8 + g) * HW + p] = s * (1.f / 64.f);
  f[((size_t)(b * 2 + 1) * 8 + g) * HW + p] = m;
}

// ---------------- Stage 3: channel attention (tiny MLP, mid ~ 2)
__global__ void channel_att_k(const float* __restrict__ avg, const float* __restrict__ mxb,
                              const float* __restrict__ w1, const float* __restrict__ w2,
                              int mid, float* __restrict__ ca) {
  int g = blockIdx.x & 7, b = blockIdx.x >> 3;
  __shared__ float red[256];
  __shared__ float hA[16], hM[16];
  for (int m = 0; m < mid; m++) {
    float sA = 0.f, sM = 0.f;
    for (int k = threadIdx.x; k < 512; k += 256) {
      int c = k >> 3, h = k & 7;
      int hp = perm_idx(g, h);
      float w = w1[m * 512 + k];
      sA += w * avg[(b * 64 + c) * 8 + hp];
      sM += w * mxb[(b * 64 + c) * 8 + hp];
    }
    red[threadIdx.x] = sA; __syncthreads();
    for (int o = 128; o > 0; o >>= 1) { if (threadIdx.x < o) red[threadIdx.x] += red[threadIdx.x + o]; __syncthreads(); }
    if (threadIdx.x == 0) hA[m] = red[0] > 0.f ? red[0] : 0.f;
    __syncthreads();
    red[threadIdx.x] = sM; __syncthreads();
    for (int o = 128; o > 0; o >>= 1) { if (threadIdx.x < o) red[threadIdx.x] += red[threadIdx.x + o]; __syncthreads(); }
    if (threadIdx.x == 0) hM[m] = red[0] > 0.f ? red[0] : 0.f;
    __syncthreads();
  }
  for (int k = threadIdx.x; k < 512; k += 256) {
    float o = 0.f;
    for (int m = 0; m < mid; m++) o += w2[k * mid + m] * (hA[m] + hM[m]);
    ca[((size_t)(b * 8 + g)) * 512 + k] = 1.f / (1.f + __expf(-o));
  }
}

// ---------------- Stage 4: spatial attention (7x7 conv over 16 maps)
__global__ void spatial_att_k(const float* __restrict__ f, const float* __restrict__ w,
                              float* __restrict__ sp) {
  int bg = blockIdx.y; int g = bg & 7; int b = bg >> 3;
  __shared__ float wk[16 * 49];
  for (int i = threadIdx.x; i < 16 * 49; i += 256) {
    int ch = i / 49, r = i % 49, kh = r / 7, kw = r % 7;
    int s = ch >> 3, j = ch & 7;
    int sh, sw; inv_spatial(g, kh, kw, 7, &sh, &sw);
    wk[i] = w[((s * 8 + perm_idx(g, j)) * 7 + sh) * 7 + sw];
  }
  __syncthreads();
  int p = blockIdx.x * 256 + threadIdx.x;
  int h = p >> 6, wc = p & 63;
  float acc = 0.f;
  for (int ch = 0; ch < 16; ch++) {
    int s = ch >> 3, j = ch & 7;
    const float* fp = f + ((size_t)(b * 2 + s) * 8 + j) * HW;
    for (int kh = 0; kh < 7; kh++) {
      int hi = h + kh - 3; if ((unsigned)hi >= 64u) continue;
      for (int kw = 0; kw < 7; kw++) {
        int wi = wc + kw - 3; if ((unsigned)wi >= 64u) continue;
        acc += fp[hi * 64 + wi] * wk[ch * 49 + kh * 7 + kw];
      }
    }
  }
  sp[(size_t)bg * HW + p] = 1.f / (1.f + __expf(-acc));
}

// ---------------- Stage 5: transformed weights baked in WMMA A-fragment order
// layout: [g][chunk=144][mtile=4][lane=32][e=16] bf16, chunk = tap*16 + cinblock
__global__ void transform_w_k(const float* __restrict__ w, unsigned short* __restrict__ wf) {
  int idx = blockIdx.x * 256 + threadIdx.x;   // 8*144*4*32*16 = 2359296
  int e = idx & 15; int t1 = idx >> 4;
  int lane = t1 & 31; t1 >>= 5;
  int mt = t1 & 3; t1 >>= 2;
  int chunk = t1 % NCHUNK; int g = t1 / NCHUNK;
  int tap = chunk >> 4, cb = chunk & 15;
  int hlf = lane >> 4, row = lane & 15;
  int cout = mt * 16 + row;
  // ISA 16-bit A 16x32 layout: lanes 0-15 K={0..7,16..23}, lanes 16-31 K={8..15,24..31}
  int K = (e < 8) ? (hlf * 8 + e) : (16 + hlf * 8 + (e - 8));
  int cin8 = cb * 32 + K;
  int cin = cin8 >> 3, gin = cin8 & 7;
  int hs = perm_idx(g, gin);
  int kh = tap / 3, kw = tap % 3, sh, sw;
  inv_spatial(g, kh, kw, 3, &sh, &sw);
  wf[idx] = f2bf(w[(((size_t)cout * 64 + cin) * 8 + hs) * 9 + sh * 3 + sw]);
}

// ---------------- Stage 6: implicit-GEMM group conv via WMMA bf16 (f32 acc)
// 256 threads = 8 wave32; tile M=64 (4 mtiles) x N=32 (2 ntiles) per (b,g);
// K staged 64 at a time (2 WMMA chunks) into double-buffered LDS in fragment order.
__global__ void __launch_bounds__(256) gconv_gemm_k(
    const unsigned short* __restrict__ yb,   // [b][pix][cin8] bf16
    const unsigned short* __restrict__ wf,   // fragment-ordered weights
    const float* __restrict__ ca,            // [b][g][512]
    const float* __restrict__ sp,            // [b][g][HW]
    const float* __restrict__ residual,      // x for final layer, else nullptr
    float* __restrict__ out)                 // [b][cout][g][HW]
{
  __shared__ __align__(32) unsigned short ldsB[2][2][2][32][16]; // [buf][chunk2][ntile][lane][e]
  int bg = blockIdx.y; int g = bg & 7; int b = bg >> 3;
  int pixbase = blockIdx.x * 32;
  int tid = threadIdx.x;
  int wid = tid >> 5, lane = tid & 31;
  int mtile = wid & 3, ntile = wid >> 2;

  const float* cap = ca + (size_t)bg * 512;
  const float* spp = sp + (size_t)bg * HW;
  const unsigned short* ybb = yb + (size_t)b * HW * CIN8;
  // per-wave A-fragment base (chunk stride = 4*512 shorts)
  const unsigned short* wfw = wf + ((size_t)g * NCHUNK * 4 + mtile) * 512 + lane * 16;

  // staging decomposition: 2048 elems (2 chunks x 2 ntiles x 32 lanes x 16) over
  // 256 threads -> 8 contiguous cin8 each (16B y-load, 2x16B ca, 16B ds-store)
  int s_c2   = tid >> 7;            // which 32-wide K chunk of the pair
  int s_nt   = (tid >> 6) & 1;
  int s_lane = (tid >> 1) & 31;
  int s_e0   = (tid & 1) * 8;
  int s_hlf  = s_lane >> 4, s_col = s_lane & 15;
  int s_pix  = pixbase + s_nt * 16 + s_col;
  int s_h = s_pix >> 6, s_w = s_pix & 63;
  unsigned short* s_dst0 = &ldsB[0][s_c2][s_nt][s_lane][s_e0];
  unsigned short* s_dst1 = &ldsB[1][s_c2][s_nt][s_lane][s_e0];
  int s_koff = s_c2 * 32 + s_hlf * 8 + s_e0 * 2;  // K offset within the 64-wide pair

  v8f acc = {0.f, 0.f, 0.f, 0.f, 0.f, 0.f, 0.f, 0.f};

  // per-tap staging state, updated only when the (uniform) tap index changes:
  // unconditional sp load from a clamped address; OOB handled by x0 mask.
  const unsigned short* st_yrow = ybb;
  float st_spv = 0.f;
  auto set_tap = [&](int tap) {
    int dh = (tap * 11) >> 5;         // == tap/3 for tap in [0,8]
    int dw = tap - dh * 3 - 1; dh -= 1;
    int hi = s_h + dh, wi = s_w + dw;
    bool valid = ((unsigned)hi < 64u) & ((unsigned)wi < 64u);
    int pin = valid ? (hi * 64 + wi) : 0;
    st_spv = spp[pin] * (valid ? 1.f : 0.f);   // branch-free masked load
    st_yrow = ybb + (size_t)pin * CIN8;
  };

  auto stage = [&](int p, int buf) {
    int cin8 = (p & 7) * 64 + s_koff;
    const unsigned short* yq = st_yrow + cin8;
    const float* cq = cap + cin8;
    unsigned short v[8];
    #pragma unroll
    for (int i = 0; i < 8; i++) v[i] = f2bf(bf2f(yq[i]) * cq[i] * st_spv);
    unsigned short* d = buf ? s_dst1 : s_dst0;
    *(uint4*)d = *(uint4*)v;
  };

  set_tap(0);
  stage(0, 0);
  for (int p = 0; p < NPAIR; p++) {
    __syncthreads();                           // stage(p) visible; consume(p-1) done
    if (p + 1 < NPAIR) {
      if (((p + 1) & 7) == 0) set_tap((p + 1) >> 3);   // uniform branch, 9 updates total
      stage(p + 1, (p + 1) & 1);
    }
    int buf = p & 1;
    const unsigned short* ap = wfw + (size_t)(p * 2) * 2048;
    v16bf a0 = *(const v16bf*)ap;
    v16bf a1 = *(const v16bf*)(ap + 2048);
    v16bf b0 = *(const v16bf*)&ldsB[buf][0][ntile][lane][0];
    v16bf b1 = *(const v16bf*)&ldsB[buf][1][ntile][lane][0];
    acc = __builtin_amdgcn_wmma_f32_16x16x32_bf16(false, a0, false, b0,
                                                  (short)0, acc, false, false);
    acc = __builtin_amdgcn_wmma_f32_16x16x32_bf16(false, a1, false, b1,
                                                  (short)0, acc, false, false);
  }

  // C/D layout: VGPR r, lanes 0-15 -> M=r, lanes 16-31 -> M=r+8; N = lane&15
  int col = lane & 15, hlf = lane >> 4;
  int pout = pixbase + ntile * 16 + col;
  if (residual) {
    #pragma unroll
    for (int r = 0; r < 8; r++) {
      int cout = mtile * 16 + r + hlf * 8;
      size_t oi = (((size_t)(b * 64 + cout)) * 8 + g) * HW + pout;
      out[oi] = acc[r] + residual[oi];
    }
  } else {
    #pragma unroll
    for (int r = 0; r < 8; r++) {
      int cout = mtile * 16 + r + hlf * 8;
      size_t oi = (((size_t)(b * 64 + cout)) * 8 + g) * HW + pout;
      out[oi] = acc[r];
    }
  }
}

extern "C" void kernel_launch(void* const* d_in, const int* in_sizes, int n_in,
                              void* d_out, int out_size, void* d_ws, size_t ws_size,
                              hipStream_t stream) {
  const float* x     = (const float*)d_in[0];
  const float* bn1g  = (const float*)d_in[1];
  const float* bn1b  = (const float*)d_in[2];
  const float* bn1m  = (const float*)d_in[3];
  const float* bn1v  = (const float*)d_in[4];
  const float* w1    = (const float*)d_in[5];
  const float* ca1w1 = (const float*)d_in[6];
  const float* ca1w2 = (const float*)d_in[7];
  const float* sa1w  = (const float*)d_in[8];
  const float* bn2g  = (const float*)d_in[9];
  const float* bn2b  = (const float*)d_in[10];
  const float* bn2m  = (const float*)d_in[11];
  const float* bn2v  = (const float*)d_in[12];
  const float* w2    = (const float*)d_in[13];
  const float* ca2w1 = (const float*)d_in[14];
  const float* ca2w2 = (const float*)d_in[15];
  const float* sa2w  = (const float*)d_in[16];
  int mid1 = in_sizes[6] / 512;
  int mid2 = in_sizes[14] / 512;

  char* ws = (char*)d_ws;
  size_t off = 0;
  auto alloc = [&](size_t bytes) -> void* {
    void* p = ws + off; off += (bytes + 255) & ~(size_t)255; return p;
  };
  unsigned short* yb  = (unsigned short*)alloc((size_t)Bb * HW * CIN8 * 2);   // 33.5 MB
  float*          avg = (float*)alloc((size_t)Bb * 64 * 8 * 4);
  float*          mxb = (float*)alloc((size_t)Bb * 64 * 8 * 4);
  float*          f   = (float*)alloc((size_t)Bb * 2 * 8 * HW * 4);           // 2 MB
  float*          cab = (float*)alloc((size_t)Bb * 8 * 512 * 4);
  float*          spb = (float*)alloc((size_t)Bb * 8 * HW * 4);               // 1 MB
  unsigned short* wfb = (unsigned short*)alloc((size_t)8 * NCHUNK * 4 * 512 * 2); // 4.7 MB
  float*          h1  = (float*)alloc((size_t)Bb * 64 * 8 * HW * 4);          // 67 MB
  (void)n_in; (void)out_size; (void)ws_size;

  // ---- layer 1
  bn_relu_stats_k<<<4096, 256, 0, stream>>>(x, bn1g, bn1b, bn1m, bn1v, yb, avg, mxb);
  ch_meanmax_k<<<1024, 256, 0, stream>>>(yb, f);
  channel_att_k<<<64, 256, 0, stream>>>(avg, mxb, ca1w1, ca1w2, mid1, cab);
  spatial_att_k<<<dim3(16, 64), 256, 0, stream>>>(f, sa1w, spb);
  transform_w_k<<<9216, 256, 0, stream>>>(w1, wfb);
  gconv_gemm_k<<<dim3(128, 64), 256, 0, stream>>>(yb, wfb, cab, spb, nullptr, h1);

  // ---- layer 2 (buffers reused; stream ordering gives the dependency)
  bn_relu_stats_k<<<4096, 256, 0, stream>>>(h1, bn2g, bn2b, bn2m, bn2v, yb, avg, mxb);
  ch_meanmax_k<<<1024, 256, 0, stream>>>(yb, f);
  channel_att_k<<<64, 256, 0, stream>>>(avg, mxb, ca2w1, ca2w2, mid2, cab);
  spatial_att_k<<<dim3(16, 64), 256, 0, stream>>>(f, sa2w, spb);
  transform_w_k<<<9216, 256, 0, stream>>>(w2, wfb);
  gconv_gemm_k<<<dim3(128, 64), 256, 0, stream>>>(yb, wfb, cab, spb, x, (float*)d_out);
}
